// NTLSTM_27504970564118
// MI455X (gfx1250) — matrix operand order repaired and checked by reading, not compiled, not run on hardware
//
#include <hip/hip_runtime.h>
#include <hip/hip_bf16.h>

// ---------------------------------------------------------------------------
// Child-Sum Tree-LSTM node_forward, bf16 WMMA + fp32 accumulate (gfx1250)
//   h_sum = sum_k h_child[:,k,:]
//   iou   = x@W_iou + h_sum@U_iou + b_iou ; i,o = sigmoid ; u = tanh
//   f_k   = sigmoid(x@W_f + h_k@U_f + b_f)
//   c     = i*u + sum_k f_k * c_child_k ;  h = o * tanh(c)
//   out   = [h ; c]  (shape [2, N, H])
// ---------------------------------------------------------------------------

typedef __attribute__((ext_vector_type(16))) __bf16 v16bf;
typedef __attribute__((ext_vector_type(8)))  float  v8f;
typedef __attribute__((ext_vector_type(8)))  unsigned int v8u;

#define D_DIM 128
#define H_DIM 128
#define K_CH  4

// fragment table offsets (each fragment = 512 bf16 = 1KB, B-layout 32x16)
#define FR_WIOU 0     // 4 ktiles x 24 ntiles = 96 frags
#define FR_UIOU 96    // 96 frags
#define FR_WF   192   // 4 x 8 = 32 frags
#define FR_UF   224   // 32 frags
#define FR_TOTAL 256
#define LDS_BYTES (FR_TOTAL * 512 * 2)  // 256 KB

static __device__ __forceinline__ v8f wmma_bf16(v16bf a, v16bf b, v8f c) {
    return __builtin_amdgcn_wmma_f32_16x16x32_bf16(
        /*neg_a=*/false, a, /*neg_b=*/false, b,
        /*c_mod=*/(short)0, c, /*reuse_a=*/false, /*reuse_b=*/false);
}

static __device__ __forceinline__ float sigmoid_fast(float v) {
    return 1.0f / (1.0f + __expf(-v));
}
static __device__ __forceinline__ float tanh_fast(float v) {
    float e = __expf(2.0f * v);
    return (e - 1.0f) / (e + 1.0f);
}

// Convert one fp32 weight matrix (128 x cols) into bf16 B-fragments in LDS.
// B-fragment layout (16-bit B, 32x16):
//   lane 0..15 : col = n0+lane,    elem e -> K = k0 + (e<8 ? e : 8 + e)
//   lane 16..31: col = n0+lane-16, elem e -> K = k0 + 8 + (e<8 ? e : 8 + e)
static __device__ void stage_weights(const float* __restrict__ src,
                                     __bf16* __restrict__ dst,
                                     int cols, int nt_stride,
                                     int tid, int nthreads) {
    const int total = 128 * cols;
    for (int idx = tid; idx < total; idx += nthreads) {
        const int k  = idx / cols;
        const int n  = idx - k * cols;
        const int kt = k >> 5, kin = k & 31;
        const int nt = n >> 4, nin = n & 15;
        const int lane = (((kin >> 3) & 1) << 4) | nin;
        const int elem = (kin & 7) | ((kin >> 4) << 3);
        dst[(kt * nt_stride + nt) * 512 + lane * 16 + elem] = (__bf16)src[idx];
    }
}

static __device__ __forceinline__ v16bf load_bfrag(const __bf16* frags,
                                                   int frag, int lane) {
    const v8u raw = *(const v8u*)(frags + frag * 512 + lane * 16); // 32B -> 2x ds_load_b128
    return __builtin_bit_cast(v16bf, raw);
}

// Load the 16 fp32 values one lane needs for an A-fragment k-tile:
// 8 consecutive at p, 8 consecutive at p+16.
static __device__ __forceinline__ void load8x2(const float* __restrict__ p,
                                               float* v) {
    const float4 t0 = ((const float4*)p)[0];
    const float4 t1 = ((const float4*)p)[1];
    const float4 t2 = ((const float4*)(p + 16))[0];
    const float4 t3 = ((const float4*)(p + 16))[1];
    v[0]=t0.x; v[1]=t0.y; v[2]=t0.z;  v[3]=t0.w;
    v[4]=t1.x; v[5]=t1.y; v[6]=t1.z;  v[7]=t1.w;
    v[8]=t2.x; v[9]=t2.y; v[10]=t2.z; v[11]=t2.w;
    v[12]=t3.x; v[13]=t3.y; v[14]=t3.z; v[15]=t3.w;
}

static __device__ __forceinline__ v16bf tobf(const float* v) {
    v16bf r;
#pragma unroll
    for (int i = 0; i < 16; ++i) r[i] = (__bf16)v[i];
    return r;
}

__global__ __launch_bounds__(256)
void NTLSTM_treelstm_wmma_kernel(const float* __restrict__ x,
                                 const float* __restrict__ h_child,
                                 const float* __restrict__ c_child,
                                 const float* __restrict__ W_iou,
                                 const float* __restrict__ U_iou,
                                 const float* __restrict__ b_iou,
                                 const float* __restrict__ W_f,
                                 const float* __restrict__ U_f,
                                 const float* __restrict__ b_f,
                                 float* __restrict__ out,
                                 int nnodes) {
    extern __shared__ __align__(32) char smem_raw[];
    __bf16* frags = (__bf16*)smem_raw;

    const int tid = threadIdx.x;
    // ---- one-time per block: swizzle weights into bf16 B-fragments in LDS
    stage_weights(W_iou, frags + FR_WIOU * 512, 3 * H_DIM, 24, tid, blockDim.x);
    stage_weights(U_iou, frags + FR_UIOU * 512, 3 * H_DIM, 24, tid, blockDim.x);
    stage_weights(W_f,   frags + FR_WF   * 512, H_DIM,     8,  tid, blockDim.x);
    stage_weights(U_f,   frags + FR_UF   * 512, H_DIM,     8,  tid, blockDim.x);
    __syncthreads();

    const int lane  = tid & 31;
    const int wid   = tid >> 5;
    const int arow  = lane & 15;          // A-fragment row (node within tile)
    const int khalf = (lane >> 4) << 3;   // A-fragment K-half select (+0 / +8)
    const int col   = lane & 15;          // C-fragment column within j-tile
    const int moff  = (lane >> 4) << 3;   // C-fragment row offset (+0 / +8)

    const int tiles   = nnodes >> 4;                 // 16-node tiles
    const int wstride = gridDim.x * (blockDim.x >> 5);
    const size_t coff = (size_t)nnodes * H_DIM;      // c-plane offset in out

    for (int t = blockIdx.x * (blockDim.x >> 5) + wid; t < tiles; t += wstride) {
        const int m0    = t << 4;
        const int nodeA = m0 + arow;
        const float* xrow = x + (size_t)nodeA * D_DIM;

        // prefetch the next tile this wave will process
        const int tn = t + wstride;
        if (tn < tiles) {
            __builtin_prefetch(x + (size_t)(tn * 16 + arow) * D_DIM, 0, 1);
            __builtin_prefetch(h_child + (size_t)(tn * 16 + arow) * (K_CH * H_DIM), 0, 1);
        }

        // ---- build A fragments: x, per-child h, and h_sum (fp32 sum -> bf16)
        v16bf a_x[4], a_hs[4], a_h[K_CH][4];
#pragma unroll
        for (int kt = 0; kt < 4; ++kt) {
            const int kbase = kt * 32 + khalf;
            float xv[16];
            load8x2(xrow + kbase, xv);
            a_x[kt] = tobf(xv);
            float hs[16];
#pragma unroll
            for (int i = 0; i < 16; ++i) hs[i] = 0.0f;
#pragma unroll
            for (int k = 0; k < K_CH; ++k) {
                const float* hrow = h_child + ((size_t)nodeA * K_CH + k) * H_DIM;
                float hv[16];
                load8x2(hrow + kbase, hv);
                a_h[k][kt] = tobf(hv);
#pragma unroll
                for (int i = 0; i < 16; ++i) hs[i] += hv[i];
            }
            a_hs[kt] = tobf(hs);
        }

        // ---- loop over 8 column tiles of H
        for (int jt = 0; jt < 8; ++jt) {
            const int j0 = jt << 4;

            v8f acc_i{}, acc_o{}, acc_u{}, xwf{};
#pragma unroll
            for (int kt = 0; kt < 4; ++kt) {
                acc_i = wmma_bf16(a_x[kt],  load_bfrag(frags, FR_WIOU + kt*24 + jt,      lane), acc_i);
                acc_i = wmma_bf16(a_hs[kt], load_bfrag(frags, FR_UIOU + kt*24 + jt,      lane), acc_i);
                acc_o = wmma_bf16(a_x[kt],  load_bfrag(frags, FR_WIOU + kt*24 + 8 + jt,  lane), acc_o);
                acc_o = wmma_bf16(a_hs[kt], load_bfrag(frags, FR_UIOU + kt*24 + 8 + jt,  lane), acc_o);
                acc_u = wmma_bf16(a_x[kt],  load_bfrag(frags, FR_WIOU + kt*24 + 16 + jt, lane), acc_u);
                acc_u = wmma_bf16(a_hs[kt], load_bfrag(frags, FR_UIOU + kt*24 + 16 + jt, lane), acc_u);
                xwf   = wmma_bf16(a_x[kt],  load_bfrag(frags, FR_WF   + kt*8 + jt,       lane), xwf);
            }

            // per-child forget gates, fused with c_child accumulation
            const float bfv = b_f[j0 + col];
            v8f csum{};
#pragma unroll
            for (int k = 0; k < K_CH; ++k) {
                v8f f = xwf;
#pragma unroll
                for (int kt = 0; kt < 4; ++kt)
                    f = wmma_bf16(a_h[k][kt], load_bfrag(frags, FR_UF + kt*8 + jt, lane), f);
#pragma unroll
                for (int r = 0; r < 8; ++r) {
                    const int node = m0 + moff + r;
                    const float cc = c_child[((size_t)node * K_CH + k) * H_DIM + j0 + col];
                    csum[r] += sigmoid_fast(f[r] + bfv) * cc;
                }
            }

            // gates + outputs
            const float bi = b_iou[j0 + col];
            const float bo = b_iou[H_DIM + j0 + col];
            const float bu = b_iou[2 * H_DIM + j0 + col];
#pragma unroll
            for (int r = 0; r < 8; ++r) {
                const int node = m0 + moff + r;
                const float iv = sigmoid_fast(acc_i[r] + bi);
                const float ov = sigmoid_fast(acc_o[r] + bo);
                const float uv = tanh_fast(acc_u[r] + bu);
                const float cv = iv * uv + csum[r];
                const float hv = ov * tanh_fast(cv);
                out[(size_t)node * H_DIM + j0 + col] = hv;          // h plane
                out[coff + (size_t)node * H_DIM + j0 + col] = cv;   // c plane
            }
        }
    }
}

extern "C" void kernel_launch(void* const* d_in, const int* in_sizes, int n_in,
                              void* d_out, int out_size, void* d_ws, size_t ws_size,
                              hipStream_t stream) {
    const float* x       = (const float*)d_in[0];
    const float* h_child = (const float*)d_in[1];
    const float* c_child = (const float*)d_in[2];
    const float* W_iou   = (const float*)d_in[3];
    const float* U_iou   = (const float*)d_in[4];
    const float* b_iou   = (const float*)d_in[5];
    const float* W_f     = (const float*)d_in[6];
    const float* U_f     = (const float*)d_in[7];
    const float* b_f     = (const float*)d_in[8];
    float* out           = (float*)d_out;

    const int nnodes = in_sizes[0] / D_DIM;  // N

    // allow 256 KB dynamic LDS (host-side attribute, not a stream op)
    (void)hipFuncSetAttribute((const void*)NTLSTM_treelstm_wmma_kernel,
                              hipFuncAttributeMaxDynamicSharedMemorySize,
                              LDS_BYTES);

    dim3 grid(256), block(256);  // persistent blocks, 8 waves each
    hipLaunchKernelGGL(NTLSTM_treelstm_wmma_kernel, grid, block, LDS_BYTES, stream,
                       x, h_child, c_child, W_iou, U_iou, b_iou, W_f, U_f, b_f,
                       out, nnodes);
}
